// TailAttention_9929964389244
// MI455X (gfx1250) — matrix operand order repaired and checked by reading, not compiled, not run on hardware
//
#include <hip/hip_runtime.h>

// ---------------------------------------------------------------------------
// MoE-routed multi-head attention for MI455X (gfx1250, wave32, WMMA bf16,
// async global->LDS copies). Compute-bound (~103 GFLOP vs ~9us HBM), so the
// design maximizes independent v_wmma_f32_16x16x32_bf16 issue per LDS access
// and hides copy latency with ASYNCcnt-tracked async-to-LDS DMA.
// ---------------------------------------------------------------------------

typedef __attribute__((ext_vector_type(16))) __bf16 v16bf;
typedef __attribute__((ext_vector_type(8)))  __bf16 v8bf;
typedef __attribute__((ext_vector_type(8)))  float  v8f;
typedef __attribute__((ext_vector_type(4)))  float  v4f;

#define B_  32
#define S_  512
#define H_  768
#define NH_ 12
#define DH_ 64
#define EC_ 4
#define EU_ 4
#define CAPACITY_ 8   // int(1.0 * 32 / 4)

__device__ __forceinline__ v16bf bf_concat(v8bf lo, v8bf hi) {
  union { v16bf v; v8bf h[2]; } u;
  u.h[0] = lo; u.h[1] = hi;
  return u.v;
}

__device__ __forceinline__ v8f wmma_bf16(v16bf a, v16bf b, v8f c) {
  // D = A(16x32 bf16) * B(32x16 bf16) + C(16x16 f32)
  return __builtin_amdgcn_wmma_f32_16x16x32_bf16(
      /*neg_a=*/false, a, /*neg_b=*/false, b,
      /*c_mod=*/(short)0, c, /*reuse_a=*/false, /*reuse_b=*/false);
}

// Async global->LDS 16B copy (CDNA5 TDM-adjacent async path, ASYNCcnt).
// Low 32 bits of a generic LDS pointer are the wave-relative LDS offset.
__device__ __forceinline__ void async_b128(unsigned lds_off, const void* g) {
  asm volatile("global_load_async_to_lds_b128 %0, %1, off"
               :: "v"(lds_off), "v"((unsigned long long)(size_t)g)
               : "memory");
}
__device__ __forceinline__ void wait_async0() {
  asm volatile("s_wait_asynccnt 0x0" ::: "memory");
}

// ---------------------------------------------------------------------------
// 1) Mean pool over the sequence axis: hm[b][h] = mean_s hidden[b][s][h]
// ---------------------------------------------------------------------------
__global__ __launch_bounds__(256)
void mean_kernel(const float* __restrict__ X, float* __restrict__ hm) {
  const int b = blockIdx.x;
  const int t = threadIdx.x;               // 256 threads, 3 columns each
  float s0 = 0.f, s1 = 0.f, s2 = 0.f;
  const float* base = X + (size_t)b * S_ * H_;
  for (int s = 0; s < S_; ++s) {
    const float* row = base + (size_t)s * H_;
    s0 += row[t];
    s1 += row[t + 256];
    s2 += row[t + 512];
  }
  const float inv = 1.0f / (float)S_;
  hm[(size_t)b * H_ + t]       = s0 * inv;
  hm[(size_t)b * H_ + t + 256] = s1 * inv;
  hm[(size_t)b * H_ + t + 512] = s2 * inv;
}

// ---------------------------------------------------------------------------
// 2) Routing: top-1 over common experts with global capacity drop; dropped
//    sequences route to unique experts.  eidx[b] in [0, EC+EU).
// ---------------------------------------------------------------------------
__global__ __launch_bounds__(32)
void route_kernel(const float* __restrict__ hm,
                  const float* __restrict__ Wsc, const float* __restrict__ bsc,
                  const float* __restrict__ Wsu, const float* __restrict__ bsu,
                  int* __restrict__ eidx) {
  __shared__ float pcmax[B_];
  __shared__ int   rc[B_];
  __shared__ int   eu[B_];
  const int b = threadIdx.x;
  if (b < B_) {
    float lc[EC_], lu[EU_];
    for (int e = 0; e < EC_; ++e) lc[e] = bsc[e];
    for (int e = 0; e < EU_; ++e) lu[e] = bsu[e];
    const float* x = hm + (size_t)b * H_;
    for (int h = 0; h < H_; ++h) {
      const float xv = x[h];
      for (int e = 0; e < EC_; ++e) lc[e] += xv * Wsc[h * EC_ + e];
      for (int e = 0; e < EU_; ++e) lu[e] += xv * Wsu[h * EU_ + e];
    }
    float mx = lc[0]; int am = 0;
    for (int e = 1; e < EC_; ++e) if (lc[e] > mx) { mx = lc[e]; am = e; }
    float se = 0.f;
    for (int e = 0; e < EC_; ++e) se += __expf(lc[e] - mx);
    pcmax[b] = 1.0f / se;                 // softmax max prob
    rc[b] = am;                           // argmax (first index on ties)
    float mu = lu[0]; int au = 0;
    for (int e = 1; e < EU_; ++e) if (lu[e] > mu) { mu = lu[e]; au = e; }
    eu[b] = EC_ + au;
  }
  __syncthreads();
  if (threadIdx.x == 0) {
    bool used[B_], kept[B_];
    int  cnt[EC_];
    for (int i = 0; i < B_; ++i) used[i] = false;
    for (int e = 0; e < EC_; ++e) cnt[e] = 0;
    for (int i = 0; i < B_; ++i) {        // stable descending selection
      int best = 0; float bv = -1e30f; bool found = false;
      for (int j = 0; j < B_; ++j)
        if (!used[j] && (!found || pcmax[j] > bv)) { bv = pcmax[j]; best = j; found = true; }
      used[best] = true;
      const int e = rc[best];
      cnt[e] += 1;
      kept[best] = (cnt[e] <= CAPACITY_);
    }
    for (int j = 0; j < B_; ++j) eidx[j] = kept[j] ? rc[j] : eu[j];
  }
}

// ---------------------------------------------------------------------------
// 3) Expert-gathered GEMM: Y[b] = X[b](S x H) @ W[eidx[b]](H x H) + bias
//    Block tile 128(M) x 64(N); 8 waves in a 4x2 grid, each computing 32x32
//    via 4 independent v_wmma_f32_16x16x32_bf16 per K-step (2 LDS b128 loads
//    per WMMA). bf16 inputs are staged via async-to-LDS DMA.
// ---------------------------------------------------------------------------
template <bool IN_BF16, bool OUT_BF16>
__global__ __launch_bounds__(256)
void moe_gemm(const void* __restrict__ Xv, const float* __restrict__ W,
              const float* __restrict__ bias, const int* __restrict__ eidx,
              void* __restrict__ Yv) {
  const int nb = blockIdx.x;     // N tile (12)
  const int mb = blockIdx.y;     // M tile (4, 128 rows each)
  const int b  = blockIdx.z;     // batch (32)
  const int e  = eidx[b];

  __shared__ alignas(16) __bf16 Xs[128][40];  // [m][k], +8 pad
  __shared__ alignas(16) __bf16 Ws[64][40];   // [n][k] (transposed)

  const int tid  = threadIdx.x;
  const int lane = tid & 31;
  const int l15  = lane & 15;
  const int lhi  = lane >> 4;          // K-half select
  const int wave = tid >> 5;
  const int mw   = wave >> 1;          // 0..3 -> rows mw*32..+32
  const int nw   = wave & 1;           // 0..1 -> cols nw*32..+32

  const int xr = tid >> 1;             // X: 2 threads/row, 16 cols each
  const int xc = (tid & 1) * 16;
  const int wk = tid >> 3;             // W: 8 threads/row(k), 8 cols(n) each
  const int wn = (tid & 7) * 8;

  const float* Wb = W + (size_t)e * H_ * H_;

  v8f acc00 = {}, acc01 = {}, acc10 = {}, acc11 = {};

  for (int kb = 0; kb < H_; kb += 32) {
    // ---- stage X tile ----
    if (IN_BF16) {
      const __bf16* X = (const __bf16*)Xv;
      const __bf16* src = X + ((size_t)b * S_ + (mb * 128 + xr)) * H_ + kb + xc;
      const unsigned l0 = (unsigned)(size_t)&Xs[xr][xc];
      async_b128(l0, src);
      async_b128(l0 + 16, src + 8);
    } else {
      const float* X = (const float*)Xv;
      const float* src = X + ((size_t)b * S_ + (mb * 128 + xr)) * H_ + kb + xc;
      v4f a = *(const v4f*)src;
      v4f c = *(const v4f*)(src + 4);
      v4f d = *(const v4f*)(src + 8);
      v4f f = *(const v4f*)(src + 12);
      v8bf x0, x1;
      x0[0] = (__bf16)a[0]; x0[1] = (__bf16)a[1]; x0[2] = (__bf16)a[2]; x0[3] = (__bf16)a[3];
      x0[4] = (__bf16)c[0]; x0[5] = (__bf16)c[1]; x0[6] = (__bf16)c[2]; x0[7] = (__bf16)c[3];
      x1[0] = (__bf16)d[0]; x1[1] = (__bf16)d[1]; x1[2] = (__bf16)d[2]; x1[3] = (__bf16)d[3];
      x1[4] = (__bf16)f[0]; x1[5] = (__bf16)f[1]; x1[6] = (__bf16)f[2]; x1[7] = (__bf16)f[3];
      *(v8bf*)&Xs[xr][xc]     = x0;
      *(v8bf*)&Xs[xr][xc + 8] = x1;
    }
    // ---- stage W tile transposed (f32 -> bf16) ----
    {
      const float* src = Wb + (size_t)(kb + wk) * H_ + nb * 64 + wn;
      v4f a = *(const v4f*)src;
      v4f c = *(const v4f*)(src + 4);
      Ws[wn + 0][wk] = (__bf16)a[0]; Ws[wn + 1][wk] = (__bf16)a[1];
      Ws[wn + 2][wk] = (__bf16)a[2]; Ws[wn + 3][wk] = (__bf16)a[3];
      Ws[wn + 4][wk] = (__bf16)c[0]; Ws[wn + 5][wk] = (__bf16)c[1];
      Ws[wn + 6][wk] = (__bf16)c[2]; Ws[wn + 7][wk] = (__bf16)c[3];
    }
    if (IN_BF16) wait_async0();
    __syncthreads();

    // A frags: rows mw*32 + {0,16} + l15; K = 16*lhi..+15 contiguous
    const int ar0 = mw * 32 + l15;
    v16bf a0 = bf_concat(*(const v8bf*)&Xs[ar0][16 * lhi],
                         *(const v8bf*)&Xs[ar0][16 * lhi + 8]);
    v16bf a1 = bf_concat(*(const v8bf*)&Xs[ar0 + 16][16 * lhi],
                         *(const v8bf*)&Xs[ar0 + 16][16 * lhi + 8]);
    // B frags: cols nw*32 + {0,16} + l15; K = 16*lhi..+15 contiguous
    const int br0  = nw * 32 + l15;
    const int koff = 16 * lhi;
    v16bf b0 = bf_concat(*(const v8bf*)&Ws[br0][koff],
                         *(const v8bf*)&Ws[br0][koff + 8]);
    v16bf b1 = bf_concat(*(const v8bf*)&Ws[br0 + 16][koff],
                         *(const v8bf*)&Ws[br0 + 16][koff + 8]);
    acc00 = wmma_bf16(a0, b0, acc00);
    acc01 = wmma_bf16(a0, b1, acc01);
    acc10 = wmma_bf16(a1, b0, acc10);
    acc11 = wmma_bf16(a1, b1, acc11);
    __syncthreads();
  }

  // ---- epilogue: bias + store (C/D layout: VGPR r -> M=r / r+8 by lane half)
  const int n0 = nb * 64 + nw * 32 + l15;
  const float bv0 = bias[(size_t)e * H_ + n0];
  const float bv1 = bias[(size_t)e * H_ + n0 + 16];
  for (int r = 0; r < 8; ++r) {
    const int m0 = mb * 128 + mw * 32 + r + 8 * lhi;
    const size_t base0 = ((size_t)b * S_ + m0) * H_;
    const size_t base1 = base0 + (size_t)16 * H_;   // m0 + 16
    if (OUT_BF16) {
      __bf16* Y = (__bf16*)Yv;
      Y[base0 + n0]      = (__bf16)(acc00[r] + bv0);
      Y[base0 + n0 + 16] = (__bf16)(acc01[r] + bv1);
      Y[base1 + n0]      = (__bf16)(acc10[r] + bv0);
      Y[base1 + n0 + 16] = (__bf16)(acc11[r] + bv1);
    } else {
      float* Y = (float*)Yv;
      Y[base0 + n0]      = acc00[r] + bv0;
      Y[base0 + n0 + 16] = acc01[r] + bv1;
      Y[base1 + n0]      = acc10[r] + bv0;
      Y[base1 + n0 + 16] = acc11[r] + bv1;
    }
  }
}

// ---------------------------------------------------------------------------
// 4) Attention: per (head, 64-query block, batch). Two-pass stable softmax.
//    K tiles are double-buffered with async-to-LDS copies so the next tile's
//    DMA overlaps the current tile's WMMAs.
// ---------------------------------------------------------------------------
__global__ __launch_bounds__(256)
void moe_attn(const __bf16* __restrict__ Q, const __bf16* __restrict__ K,
              const __bf16* __restrict__ V, const float* __restrict__ mask,
              __bf16* __restrict__ CTX) {
  const int head = blockIdx.x;   // 12
  const int qb   = blockIdx.y;   // 8
  const int b    = blockIdx.z;   // 32
  const int c0   = head * DH_;

  __shared__ alignas(16) __bf16 Qs[64][72];
  __shared__ alignas(16) __bf16 Ks[2][64][72];  // double-buffered
  __shared__ alignas(16) __bf16 Ps[64][72];
  __shared__ alignas(16) __bf16 Vt[64][72];     // [d][t]
  __shared__ alignas(16) float  Ss[64][68];
  __shared__ float  mrow[64], lrow[64];

  const int tid  = threadIdx.x;
  const int lane = tid & 31;
  const int l15  = lane & 15;
  const int lhi  = lane >> 4;
  const int wave = tid >> 5;
  const int mw   = wave >> 1;    // 0..3
  const int nw   = wave & 1;     // 0..1

  const int tr = tid >> 2;             // 4 threads/row, 16 elems each
  const int tc = (tid & 3) * 16;

  const float sc = 0.125f;  // 1/sqrt(64)

  // async copy of one 64x64 bf16 K tile into buffer `buf`
  auto issue_K = [&](int kb, int buf) {
    const __bf16* src = K + ((size_t)b * S_ + (kb * 64 + tr)) * H_ + c0 + tc;
    const unsigned l0 = (unsigned)(size_t)&Ks[buf][tr][tc];
    async_b128(l0, src);
    async_b128(l0 + 16, src + 8);
  };

  // ---- stage Q tile (async) + K[0]; init softmax stats ----
  {
    const __bf16* src = Q + ((size_t)b * S_ + (qb * 64 + tr)) * H_ + c0 + tc;
    const unsigned l0 = (unsigned)(size_t)&Qs[tr][tc];
    async_b128(l0, src);
    async_b128(l0 + 16, src + 8);
  }
  issue_K(0, 0);
  if (tid < 64) { mrow[tid] = -1e30f; lrow[tid] = 0.f; }

  // score tile for buffer `buf` -> Ss (scaled + mask)
  auto score_tile = [&](int kb, int buf) {
    v8f s0 = {}; v8f s1 = {};
    for (int kk = 0; kk < DH_; kk += 32) {
      const int arow = mw * 16 + l15;
      v16bf af = bf_concat(*(const v8bf*)&Qs[arow][kk + 16 * lhi],
                           *(const v8bf*)&Qs[arow][kk + 16 * lhi + 8]);
      const int br0  = nw * 32 + l15;
      const int koff = kk + 16 * lhi;
      v16bf bf0 = bf_concat(*(const v8bf*)&Ks[buf][br0][koff],
                            *(const v8bf*)&Ks[buf][br0][koff + 8]);
      v16bf bf1 = bf_concat(*(const v8bf*)&Ks[buf][br0 + 16][koff],
                            *(const v8bf*)&Ks[buf][br0 + 16][koff + 8]);
      s0 = wmma_bf16(af, bf0, s0);
      s1 = wmma_bf16(af, bf1, s1);
    }
    const int nl0 = nw * 32 + l15;
    const float mk0 = mask[(size_t)b * S_ + kb * 64 + nl0];
    const float mk1 = mask[(size_t)b * S_ + kb * 64 + nl0 + 16];
    for (int r = 0; r < 8; ++r) {
      const int m = mw * 16 + r + 8 * lhi;
      Ss[m][nl0]      = s0[r] * sc + mk0;
      Ss[m][nl0 + 16] = s1[r] * sc + mk1;
    }
  };

  // ================= pass 1: row max / row sum =================
  for (int kb = 0; kb < S_ / 64; ++kb) {
    const int buf = kb & 1;
    wait_async0();
    __syncthreads();                          // K[buf] (and Q) visible
    if (kb + 1 < S_ / 64) issue_K(kb + 1, buf ^ 1);  // overlap DMA with WMMA
    score_tile(kb, buf);
    __syncthreads();
    if (tid < 64) {
      const float mo = mrow[tid];
      float tmax = -1e30f;
      for (int j = 0; j < 64; ++j) tmax = fmaxf(tmax, Ss[tid][j]);
      const float mn = fmaxf(mo, tmax);
      float acc = lrow[tid] * __expf(mo - mn);
      for (int j = 0; j < 64; ++j) acc += __expf(Ss[tid][j] - mn);
      mrow[tid] = mn;
      lrow[tid] = acc;
    }
    __syncthreads();
  }

  // ================= pass 2: ctx = P @ V =================
  issue_K(0, 0);
  v8f c00 = {}, c01 = {};
  for (int kb = 0; kb < S_ / 64; ++kb) {
    const int buf = kb & 1;
    // stage V tile transposed (manual: needs transpose)
    {
      const __bf16* vsrc = V + ((size_t)b * S_ + (kb * 64 + tr)) * H_ + c0 + tc;
      v8bf v0 = *(const v8bf*)vsrc;
      v8bf v1 = *(const v8bf*)(vsrc + 8);
      for (int i = 0; i < 8; ++i) Vt[tc + i][tr]     = v0[i];
      for (int i = 0; i < 8; ++i) Vt[tc + 8 + i][tr] = v1[i];
    }
    wait_async0();
    __syncthreads();                          // K[buf] + Vt visible
    if (kb + 1 < S_ / 64) issue_K(kb + 1, buf ^ 1);
    score_tile(kb, buf);
    __syncthreads();
    // P = exp(S - m) / l  (bf16)
    {
      const float mm  = mrow[tr];
      const float inv = 1.0f / lrow[tr];
      for (int i = 0; i < 16; ++i)
        Ps[tr][tc + i] = (__bf16)(__expf(Ss[tr][tc + i] - mm) * inv);
    }
    __syncthreads();
    // ctx += P(64x64) @ V(64x64): A rows = queries, B rows (Vt) = d
    for (int kk = 0; kk < 64; kk += 32) {
      const int arow = mw * 16 + l15;
      v16bf af = bf_concat(*(const v8bf*)&Ps[arow][kk + 16 * lhi],
                           *(const v8bf*)&Ps[arow][kk + 16 * lhi + 8]);
      const int br0  = nw * 32 + l15;
      const int koff = kk + 16 * lhi;
      v16bf bf0 = bf_concat(*(const v8bf*)&Vt[br0][koff],
                            *(const v8bf*)&Vt[br0][koff + 8]);
      v16bf bf1 = bf_concat(*(const v8bf*)&Vt[br0 + 16][koff],
                            *(const v8bf*)&Vt[br0 + 16][koff + 8]);
      c00 = wmma_bf16(af, bf0, c00);
      c01 = wmma_bf16(af, bf1, c01);
    }
    __syncthreads();
  }

  // ---- store ctx tile (bf16, [B,S,H] with head cols) ----
  const int n0 = nw * 32 + l15;
  for (int r = 0; r < 8; ++r) {
    const int m = qb * 64 + mw * 16 + r + 8 * lhi;
    const size_t base = ((size_t)b * S_ + m) * H_ + c0;
    CTX[base + n0]      = (__bf16)c00[r];
    CTX[base + n0 + 16] = (__bf16)c01[r];
  }
}

// ---------------------------------------------------------------------------
// launcher
// ---------------------------------------------------------------------------
extern "C" void kernel_launch(void* const* d_in, const int* in_sizes, int n_in,
                              void* d_out, int out_size, void* d_ws, size_t ws_size,
                              hipStream_t stream) {
  const float* hs   = (const float*)d_in[0];   // [B,S,H]
  const float* msk  = (const float*)d_in[1];   // [B,S]
  const float* Wq   = (const float*)d_in[2];
  const float* bq   = (const float*)d_in[3];
  const float* Wk   = (const float*)d_in[4];
  const float* bk   = (const float*)d_in[5];
  const float* Wv   = (const float*)d_in[6];
  const float* bv   = (const float*)d_in[7];
  const float* Wo   = (const float*)d_in[8];
  const float* bo   = (const float*)d_in[9];
  const float* Wsc  = (const float*)d_in[10];
  const float* bsc  = (const float*)d_in[11];
  const float* Wsu  = (const float*)d_in[12];
  const float* bsu  = (const float*)d_in[13];

  char* ws = (char*)d_ws;
  int*   eidx = (int*)ws;                                    // 32 ints
  float* hm   = (float*)(ws + 256);                          // 32*768 f32
  const size_t NELT = (size_t)B_ * S_ * H_;                  // 12.58M
  __bf16* qbuf   = (__bf16*)(ws + 256 + (size_t)B_ * H_ * 4);
  __bf16* kbuf   = qbuf + NELT;
  __bf16* vbuf   = kbuf + NELT;
  __bf16* ctxbuf = vbuf + NELT;

  mean_kernel<<<B_, 256, 0, stream>>>(hs, hm);
  route_kernel<<<1, 32, 0, stream>>>(hm, Wsc, bsc, Wsu, bsu, eidx);

  dim3 gg(H_ / 64, S_ / 128, B_);    // (12, 4, 32)
  moe_gemm<false, true><<<gg, 256, 0, stream>>>(hs, Wq, bq, eidx, qbuf);
  moe_gemm<false, true><<<gg, 256, 0, stream>>>(hs, Wk, bk, eidx, kbuf);
  moe_gemm<false, true><<<gg, 256, 0, stream>>>(hs, Wv, bv, eidx, vbuf);

  dim3 ga(NH_, S_ / 64, B_);         // (12, 8, 32)
  moe_attn<<<ga, 256, 0, stream>>>(qbuf, kbuf, vbuf, msk, ctxbuf);

  // final projection; scale == p/stop_grad(p) == 1.0 in forward, so omitted
  moe_gemm<true, false><<<gg, 256, 0, stream>>>(ctxbuf, Wo, bo, eidx, d_out);
}
